// Qwen3MoeHeterogeneousSparseMoeBlock_90117003804877
// MI455X (gfx1250) — matrix operand order repaired and checked by reading, not compiled, not run on hardware
//
#include <hip/hip_runtime.h>
#include <hip/hip_bf16.h>

#define T_  2048
#define D_  1024
#define E_  16
#define F_  512
#define MT_ 32        // tokens per FFN tile (two 16-row WMMA sub-tiles)

typedef __attribute__((ext_vector_type(16))) __bf16 v16bf;
typedef __attribute__((ext_vector_type(8)))  __bf16 v8bf;
typedef __attribute__((ext_vector_type(4)))  __bf16 v4bf;
typedef __attribute__((ext_vector_type(8)))  float  v8f;
typedef __attribute__((ext_vector_type(4)))  float  f4;

__device__ __forceinline__ __bf16 tobf(float x) { return (__bf16)x; }

// ---------------------------------------------------------------------------
// Router: one wave per token; butterfly-reduced logits; analytic top-2 renorm
// (softmax denominator cancels under NORM_TOPK): w1 = 1/(1+exp(l2-l1)).
// ---------------------------------------------------------------------------
__global__ __launch_bounds__(256) void moe_router(
    const float* __restrict__ hidden, const float* __restrict__ wg,
    int* __restrict__ cnt, int* __restrict__ elist, float* __restrict__ ewt)
{
    const int lane = threadIdx.x & 31;
    const int wv   = threadIdx.x >> 5;
    const int t    = blockIdx.x * 8 + wv;

    const float* hrow = hidden + (size_t)t * D_;
    float hv[32];
#pragma unroll
    for (int i = 0; i < 32; ++i) hv[i] = hrow[i * 32 + lane];

    float logit[E_];
#pragma unroll
    for (int e = 0; e < E_; ++e) {
        const float* wr = wg + (size_t)e * D_;
        float acc = 0.f;
#pragma unroll
        for (int i = 0; i < 32; ++i) acc += hv[i] * wr[i * 32 + lane];
#pragma unroll
        for (int off = 16; off > 0; off >>= 1)
            acc += __shfl_xor(acc, off, 32);
        logit[e] = acc;
    }

    int i1 = 0; float l1 = logit[0];
#pragma unroll
    for (int e = 1; e < E_; ++e) if (logit[e] > l1) { l1 = logit[e]; i1 = e; }
    int i2 = -1; float l2 = -3.4e38f;
#pragma unroll
    for (int e = 0; e < E_; ++e) if (e != i1 && logit[e] > l2) { l2 = logit[e]; i2 = e; }

    const float r  = __expf(l2 - l1);
    const float w1 = 1.f / (1.f + r);
    const float w2 = r * w1;

    if (lane == 0) {
        int p1 = atomicAdd(&cnt[i1], 1);
        elist[i1 * T_ + p1] = t;  ewt[i1 * T_ + p1] = w1;
        int p2 = atomicAdd(&cnt[i2], 1);
        elist[i2 * T_ + p2] = t;  ewt[i2 * T_ + p2] = w2;
    }
}

// ---------------------------------------------------------------------------
// Weight conversion + swizzle: fp32 row-major [R, K] -> bf16 WMMA-B fragments.
// Fragment (row-tile t of 16 rows, k-tile kt of 32): 512 bf16 (1 KB) record,
// lane l<16 -> col t*16+l, k = kt*32+0..15 ; lane l+16 -> k = kt*32+16..31.
// One wave per fragment; 8 waves per block cover 8 consecutive k-tiles.
// ---------------------------------------------------------------------------
__global__ __launch_bounds__(256) void conv_b(
    const float* __restrict__ src, __bf16* __restrict__ dst, int K)
{
    const int lane = threadIdx.x & 31;
    const int wv   = threadIdx.x >> 5;
    const int gk   = K >> 8;                      // k-tile groups of 8 per row-tile
    const int t    = blockIdx.x / gk;             // 16-row tile
    const int kt   = (blockIdx.x % gk) * 8 + wv;  // 32-k tile
    const int row  = t * 16 + (lane & 15);
    const int k0   = kt * 32 + ((lane >= 16) ? 16 : 0);

    const float* p = src + (size_t)row * K + k0;
    const f4 a = *(const f4*)(p);
    const f4 b = *(const f4*)(p + 4);
    const f4 c = *(const f4*)(p + 8);
    const f4 d = *(const f4*)(p + 12);
    v8bf lo, hi;
#pragma unroll
    for (int i = 0; i < 4; ++i) {
        lo[i] = tobf(a[i]); lo[4 + i] = tobf(b[i]);
        hi[i] = tobf(c[i]); hi[4 + i] = tobf(d[i]);
    }
    __bf16* q = dst + ((size_t)t * (K >> 5) + kt) * 512 + lane * 16;
    *(v8bf*)(q)     = lo;
    *(v8bf*)(q + 8) = hi;
}

// ---------------------------------------------------------------------------
// Grouped expert FFN, 32 tokens per block. Each wave owns 128 output columns
// (8 x 16-col WMMA tiles) x 2 row sub-tiles; every B fragment feeds 2 WMMAs.
// B comes pre-converted/pre-swizzled bf16 (two coalesced b128 loads per lane,
// zero cvts in the hot loop). LDS 64KB union: hid (32x1024 bf16, XOR-swizzled
// 16B granules) -> silu(gate) 32x512 linear | act 32x512 swizzled.
// ---------------------------------------------------------------------------
__global__ __launch_bounds__(256) void moe_ffn(
    const float*  __restrict__ hidden,
    const __bf16* __restrict__ w13b,   // [E*64 tiles][32 kt][512]
    const __bf16* __restrict__ w2b,    // [E*64 tiles][16 kt][512]
    const int*    __restrict__ cnt,
    const int*    __restrict__ elist,
    const float*  __restrict__ ewt,
    float*        __restrict__ out)
{
    const int e    = blockIdx.y;
    const int n    = cnt[e];
    const int base = blockIdx.x * MT_;
    if (base >= n) return;
    const int nv = (n - base < MT_) ? (n - base) : MT_;

    __shared__ __align__(16) unsigned char smem[65536];
    __bf16* hid = (__bf16*)smem;            // phase 1: 32 x 1024 (swizzled)
    __bf16* sg  = (__bf16*)smem;            // phase 2: silu(gate) 32 x 512 (linear)
    __bf16* act = (__bf16*)(smem + 32768);  // phase 2: act 32 x 512 (swizzled)

    const int tid  = threadIdx.x;
    const int lane = tid & 31;
    const int wv   = tid >> 5;
    const int ln15 = lane & 15;
    const int hi8  = (lane >> 4) * 8;        // D-row offset; A-frag K offset is 2*hi8 (contiguous k = 16h + s)

    const int*   el = elist + e * T_ + base;
    const float* ew = ewt   + e * T_ + base;

    // ---- Gather 32 hidden rows -> LDS bf16, 16B-granule XOR swizzle --------
    {
        const int c = tid * 4, gr = c >> 3, cw = c & 7;
        for (int m = 0; m < MT_; ++m) {             // uniform loop, scalar el[m]
            f4 v = {0.f, 0.f, 0.f, 0.f};
            if (m < nv) v = *(const f4*)(hidden + (size_t)el[m] * D_ + c);
            v4bf o;
#pragma unroll
            for (int i = 0; i < 4; ++i) o[i] = tobf(v[i]);
            *(v4bf*)(hid + m * D_ + ((((gr ^ (m & 15)) << 3)) | cw)) = o;
        }
    }
    __syncthreads();

    // ---------------- GEMM1: h[32,1024] = X[32,1024] @ w13[e]^T -------------
    v8f acc0[8], acc1[8];
#pragma unroll
    for (int j = 0; j < 8; ++j) {
        acc0[j] = (v8f){0.f,0.f,0.f,0.f,0.f,0.f,0.f,0.f};
        acc1[j] = (v8f){0.f,0.f,0.f,0.f,0.f,0.f,0.f,0.f};
    }

    const int cb = wv * 128;
    // fragment records: tile (e*64 + cb/16 + j), kt, lane*16
    const __bf16* bb1 = w13b + ((size_t)(e * 64 + (cb >> 4)) * 32) * 512 + lane * 16;
    const __bf16* ar0 = hid + ln15 * D_;
    const __bf16* ar1 = hid + (ln15 + 16) * D_;

    for (int kt = 0; kt < 32; ++kt) {
        const int o1 = (((((kt * 32 + 2 * hi8) >> 3) ^ ln15)) << 3);
        const int o2 = (((((kt * 32 + 8 + 2 * hi8) >> 3) ^ ln15)) << 3);
        v16bf a0, a1;
        {
            const v8bf x0 = *(const v8bf*)(ar0 + o1);
            const v8bf x1 = *(const v8bf*)(ar0 + o2);
            const v8bf y0 = *(const v8bf*)(ar1 + o1);
            const v8bf y1 = *(const v8bf*)(ar1 + o2);
#pragma unroll
            for (int i = 0; i < 8; ++i) {
                a0[i] = x0[i]; a0[8 + i] = x1[i];
                a1[i] = y0[i]; a1[8 + i] = y1[i];
            }
        }
#pragma unroll
        for (int j = 0; j < 8; ++j) {
            const __bf16* p = bb1 + (size_t)j * (32 * 512) + (size_t)kt * 512;
            const v8bf blo = *(const v8bf*)(p);
            const v8bf bhi = *(const v8bf*)(p + 8);
            v16bf b;
#pragma unroll
            for (int i = 0; i < 8; ++i) { b[i] = blo[i]; b[8 + i] = bhi[i]; }
            acc0[j] = __builtin_amdgcn_wmma_f32_16x16x32_bf16(
                          false, a0, false, b, (short)0, acc0[j], false, false);
            acc1[j] = __builtin_amdgcn_wmma_f32_16x16x32_bf16(
                          false, a1, false, b, (short)0, acc1[j], false, false);
        }
    }
    __syncthreads();   // all GEMM1 LDS reads done before overwriting hid

    // Waves 0..3 hold gate (cols 0..511): park silu(gate) as bf16 (linear)
    if (wv < 4) {
        const int c0 = wv * 128;
#pragma unroll
        for (int j = 0; j < 8; ++j)
#pragma unroll
            for (int r = 0; r < 8; ++r) {
                const int f = c0 + j * 16 + ln15;
                const float g0  = acc0[j][r];
                const float g1v = acc1[j][r];
                sg[(r + hi8) * 512 + f]      = tobf(g0 * (1.f / (1.f + __expf(-g0))));
                sg[(16 + r + hi8) * 512 + f] = tobf(g1v * (1.f / (1.f + __expf(-g1v))));
            }
    }
    __syncthreads();

    // Waves 4..7 hold up (cols 512..1023): act = silu(gate) * up -> swizzled bf16
    if (wv >= 4) {
        const int c0 = (wv - 4) * 128;
#pragma unroll
        for (int j = 0; j < 8; ++j)
#pragma unroll
            for (int r = 0; r < 8; ++r) {
                const int f = c0 + j * 16 + ln15;
                {
                    const int m = r + hi8;
                    const float a = (float)sg[m * 512 + f] * acc0[j][r];
                    act[m * 512 + (((((f >> 3) ^ (m & 15)) << 3)) | (f & 7))] = tobf(a);
                }
                {
                    const int m = 16 + r + hi8;
                    const float a = (float)sg[m * 512 + f] * acc1[j][r];
                    act[m * 512 + (((((f >> 3) ^ (m & 15)) << 3)) | (f & 7))] = tobf(a);
                }
            }
    }
    __syncthreads();

    // ---------------- GEMM2: y[32,1024] = act[32,512] @ w2[e]^T -------------
    v8f d0[8], d1[8];
#pragma unroll
    for (int j = 0; j < 8; ++j) {
        d0[j] = (v8f){0.f,0.f,0.f,0.f,0.f,0.f,0.f,0.f};
        d1[j] = (v8f){0.f,0.f,0.f,0.f,0.f,0.f,0.f,0.f};
    }

    const __bf16* bb2 = w2b + ((size_t)(e * 64 + (cb >> 4)) * 16) * 512 + lane * 16;
    const __bf16* ac0 = act + ln15 * 512;
    const __bf16* ac1 = act + (ln15 + 16) * 512;

    for (int kt = 0; kt < 16; ++kt) {
        const int o1 = (((((kt * 32 + 2 * hi8) >> 3) ^ ln15)) << 3);
        const int o2 = (((((kt * 32 + 8 + 2 * hi8) >> 3) ^ ln15)) << 3);
        v16bf a0, a1;
        {
            const v8bf x0 = *(const v8bf*)(ac0 + o1);
            const v8bf x1 = *(const v8bf*)(ac0 + o2);
            const v8bf y0 = *(const v8bf*)(ac1 + o1);
            const v8bf y1 = *(const v8bf*)(ac1 + o2);
#pragma unroll
            for (int i = 0; i < 8; ++i) {
                a0[i] = x0[i]; a0[8 + i] = x1[i];
                a1[i] = y0[i]; a1[8 + i] = y1[i];
            }
        }
#pragma unroll
        for (int j = 0; j < 8; ++j) {
            const __bf16* p = bb2 + (size_t)j * (16 * 512) + (size_t)kt * 512;
            const v8bf blo = *(const v8bf*)(p);
            const v8bf bhi = *(const v8bf*)(p + 8);
            v16bf b;
#pragma unroll
            for (int i = 0; i < 8; ++i) { b[i] = blo[i]; b[8 + i] = bhi[i]; }
            d0[j] = __builtin_amdgcn_wmma_f32_16x16x32_bf16(
                        false, a0, false, b, (short)0, d0[j], false, false);
            d1[j] = __builtin_amdgcn_wmma_f32_16x16x32_bf16(
                        false, a1, false, b, (short)0, d1[j], false, false);
        }
    }

    // Weighted scatter. Each out element receives exactly 2 atomic adds onto
    // an exact zero -> order-independent (bit-deterministic across replays).
#pragma unroll
    for (int r = 0; r < 8; ++r) {
        {
            const int m = r + hi8;
            if (m < nv) {
                const float w = ew[m];
                float* orow = out + (size_t)el[m] * D_;
#pragma unroll
                for (int j = 0; j < 8; ++j)
                    unsafeAtomicAdd(orow + cb + j * 16 + ln15, w * d0[j][r]);
            }
        }
        {
            const int m = 16 + r + hi8;
            if (m < nv) {
                const float w = ew[m];
                float* orow = out + (size_t)el[m] * D_;
#pragma unroll
                for (int j = 0; j < 8; ++j)
                    unsafeAtomicAdd(orow + cb + j * 16 + ln15, w * d1[j][r]);
            }
        }
    }
}

// ---------------------------------------------------------------------------
extern "C" void kernel_launch(void* const* d_in, const int* in_sizes, int n_in,
                              void* d_out, int out_size, void* d_ws, size_t ws_size,
                              hipStream_t stream) {
    const float* hidden = (const float*)d_in[0];   // [T, D]
    const float* wgate  = (const float*)d_in[1];   // [E, D]
    const float* w13    = (const float*)d_in[2];   // [E, 2F, D]
    const float* w2     = (const float*)d_in[3];   // [E, D, F]
    float* out = (float*)d_out;                    // [T, D]

    unsigned char* ws = (unsigned char*)d_ws;
    int*    cnt   = (int*)ws;                              // offset 0 (1 KB slot)
    int*    elist = (int*)(ws + 1024);                     // E*T ids   (128 KB)
    float*  ewt   = (float*)(ws + 1024 + 4 * E_ * T_);     // E*T wts   (128 KB)
    __bf16* w13b  = (__bf16*)(ws + 263168);                // 32 MB bf16 frags
    __bf16* w2b   = (__bf16*)(ws + 263168 + 33554432);     // 16 MB bf16 frags

    hipMemsetAsync(cnt, 0, 64, stream);
    hipMemsetAsync(out, 0, (size_t)T_ * D_ * sizeof(float), stream);

    // One-pass weight convert+swizzle: rows = E*2F (w13), E*D (w2)
    conv_b<<<(E_ * 2 * F_ / 16) * (D_ / 256), 256, 0, stream>>>(w13, w13b, D_);
    conv_b<<<(E_ * D_ / 16) * (F_ / 256), 256, 0, stream>>>(w2, w2b, F_);

    moe_router<<<T_ / 8, 256, 0, stream>>>(hidden, wgate, cnt, elist, ewt);

    dim3 grid(T_ / MT_, E_, 1);   // max 64 tiles per expert; extras exit early
    moe_ffn<<<grid, 256, 0, stream>>>(hidden, w13b, w2b, cnt, elist, ewt, out);
}
